// Attention_77146202570808
// MI455X (gfx1250) — matrix operand order repaired and checked by reading, not compiled, not run on hardware
//
#include <hip/hip_runtime.h>

// ---------------------------------------------------------------------------
// MI455X (gfx1250) cross-attention block.
// Compute-bound (~108 GFLOP vs ~0.1-0.2 GB traffic at 23.3 TB/s): all GEMMs
// and both attention matmuls run on v_wmma_f32_16x16x32_f16 (f16 operands
// staged through LDS, f32 accumulate). Flash-style attention avoids the
// 402 MB score tensor. Softmax reductions are pure-VALU DPP16 butterflies
// (no ds_bpermute), and all per-row state is force-unrolled into registers
// (no v_movrels dynamic VGPR indexing).
// ---------------------------------------------------------------------------

typedef __attribute__((ext_vector_type(16))) _Float16 v16h;
typedef __attribute__((ext_vector_type(8)))  float    v8f;

constexpr int Bc = 8, Pc = 1024, Mc = 256, Dc = 768, Hc = 12, DHc = 64;

static __device__ __forceinline__ v8f wmma_f16(v16h a, v16h b, v8f c) {
  return __builtin_amdgcn_wmma_f32_16x16x32_f16(false, a, false, b,
                                                (short)0, c, false, false);
}

// ---- 16-lane butterfly reductions via DPP16 (VALU only, wave32 rows) ------
// quad_perm(1,0,3,2)=0xB1 -> xor1 ; quad_perm(2,3,0,1)=0x4E -> xor2 ;
// row_half_mirror=0x141 combines quads within 8 ; row_mirror=0x140 combines
// the two 8-groups within each 16-lane row. Lanes 0-15 / 16-31 stay separate.
template<int CTRL>
static __device__ __forceinline__ float dpp_maxstep(float x) {
  int o = __builtin_amdgcn_update_dpp(0, __float_as_int(x), CTRL, 0xF, 0xF, true);
  return fmaxf(x, __int_as_float(o));
}
template<int CTRL>
static __device__ __forceinline__ float dpp_addstep(float x) {
  int o = __builtin_amdgcn_update_dpp(0, __float_as_int(x), CTRL, 0xF, 0xF, true);
  return x + __int_as_float(o);
}
static __device__ __forceinline__ float red_max16(float x) {
  x = dpp_maxstep<0xB1>(x); x = dpp_maxstep<0x4E>(x);
  x = dpp_maxstep<0x141>(x); x = dpp_maxstep<0x140>(x);
  return x;
}
static __device__ __forceinline__ float red_sum16(float x) {
  x = dpp_addstep<0xB1>(x); x = dpp_addstep<0x4E>(x);
  x = dpp_addstep<0x141>(x); x = dpp_addstep<0x140>(x);
  return x;
}

// ===========================================================================
// Generic batched GEMM:  C[b] = alpha * opA(A[b]) @ B[b]  (+ bias)
//   TA=0: A is [Mdim,K] row-major ; TA=1: A_gemm[m,k] = Amem[k*lda + m]
//   BIAS: 0 none, 1 per-column (Linear bias), 2 per-row (fc bias over seq)
// Tile: 128(M) x 64(N) x 32(K); 8 waves, each owns a 32x32 C block.
// All problem dims are exact tile multiples -> no guards; EXEC all-ones
// around WMMA per ISA requirement.
// ===========================================================================
template<int TA, int BIAS>
__global__ __launch_bounds__(256)
void gemm_wmma(const float* __restrict__ A, int lda, long long strideA,
               const float* __restrict__ Bm, int ldb, long long strideB,
               const float* __restrict__ bias,
               float* __restrict__ C, int ldc, long long strideC,
               int Mdim, int Ndim, int Kdim, float alpha)
{
  (void)Mdim; (void)Ndim;
  constexpr int BM = 128, BN = 64, KT = 32, LDS_K = KT + 8; // +8 halves: 16B align
  __shared__ _Float16 As[BM][LDS_K];
  __shared__ _Float16 Bs[BN][LDS_K];

  const float* Ab = A  + (size_t)blockIdx.z * strideA;
  const float* Bb = Bm + (size_t)blockIdx.z * strideB;
  float*       Cb = C  + (size_t)blockIdx.z * strideC;

  const int m0   = blockIdx.y * BM;
  const int n0   = blockIdx.x * BN;
  const int tid  = threadIdx.x;
  const int lane = tid & 31;
  const int w    = tid >> 5;
  const int mw   = (w & 3) * 32;
  const int nw   = (w >> 2) * 32;
  const int half = (lane >> 4) & 1;
  const int lr   = lane & 15;

  v8f acc[2][2] = {};

  for (int k0 = 0; k0 < Kdim; k0 += KT) {
    __syncthreads();
    // ---- stage A tile (f32 -> f16), coalesced for either layout ----
    for (int i = tid; i < BM * KT; i += 256) {
      if (TA) { int r = i & (BM - 1), kk = i >> 7;
        As[r][kk] = (_Float16)Ab[(size_t)(k0 + kk) * lda + (m0 + r)];
      } else {  int r = i >> 5, kk = i & 31;
        As[r][kk] = (_Float16)Ab[(size_t)(m0 + r) * lda + (k0 + kk)];
      }
    }
    // ---- stage B tile [K,N] -> Bs[n][k] (coalesced reads over n) ----
    for (int i = tid; i < BN * KT; i += 256) {
      int nn = i & (BN - 1), kk = i >> 6;
      Bs[nn][kk] = (_Float16)Bb[(size_t)(k0 + kk) * ldb + (n0 + nn)];
    }
    // ---- prefetch next K tile into cache (global_prefetch_b8) ----
    if (k0 + KT < Kdim) {
      if (TA) __builtin_prefetch(&Ab[(size_t)(k0 + KT + (tid >> 7)) * lda + m0 + (tid & 127)], 0, 1);
      else    __builtin_prefetch(&Ab[(size_t)(m0 + (tid >> 5)) * lda + k0 + KT + (tid & 31)], 0, 1);
      __builtin_prefetch(&Bb[(size_t)(k0 + KT + (tid >> 6)) * ldb + n0 + (tid & 63)], 0, 1);
    }
    __syncthreads();

    // ---- fragment loads (2x ds_load_b128 each, ISA 7.12.2 layout) ----
    v16h afrag[2], bfrag[2];
#pragma unroll
    for (int i = 0; i < 2; ++i) {
      const _Float16* p = &As[mw + i * 16 + lr][half * 8];
      ((uint4*)&afrag[i])[0] = *(const uint4*)p;        // k 0..7   (or 8..15)
      ((uint4*)&afrag[i])[1] = *(const uint4*)(p + 16); // k 16..23 (or 24..31)
    }
#pragma unroll
    for (int j = 0; j < 2; ++j) {
      const _Float16* p = &Bs[nw + j * 16 + lr][half * 8];
      ((uint4*)&bfrag[j])[0] = *(const uint4*)p;        // k 0..7   (or 8..15)
      ((uint4*)&bfrag[j])[1] = *(const uint4*)(p + 16); // k 16..23 (or 24..31)
    }
#pragma unroll
    for (int i = 0; i < 2; ++i)
#pragma unroll
      for (int j = 0; j < 2; ++j)
        acc[i][j] = wmma_f16(afrag[i], bfrag[j], acc[i][j]);
  }

  // ---- epilogue: lane(0-15)->N=lane,M=r ; lane(16-31)->M=8+r ----
#pragma unroll
  for (int i = 0; i < 2; ++i)
#pragma unroll
    for (int j = 0; j < 2; ++j) {
      int mb = m0 + mw + i * 16 + 8 * half;
      int n  = n0 + nw + j * 16 + lr;
#pragma unroll
      for (int r = 0; r < 8; ++r) {
        float vv = alpha * acc[i][j][r];
        if (BIAS == 1) vv += bias[n];
        if (BIAS == 2) vv += bias[mb + r];
        Cb[(size_t)(mb + r) * ldc + n] = vv;
      }
    }
}

// ===========================================================================
// Flash attention, one (batch, head, 64-row Q tile) per block.
// 4 waves x 16 query rows; 32-key tiles in LDS; online softmax with DPP16
// butterfly reductions; probs re-fragmented through per-wave LDS.
// ===========================================================================
__global__ __launch_bounds__(128)
void flash_attn(const float* __restrict__ Q, const float* __restrict__ K,
                const float* __restrict__ V,
                float* __restrict__ Out, long long outStride, int rowOff,
                int Sq, int Sk, float scale)
{
  __shared__ _Float16 Qs[64][72];       // Q tile, [row][dh]
  __shared__ _Float16 Ks[32][72];       // K tile, [key][dh]
  __shared__ _Float16 Vs[64][40];       // V tile transposed, [dh][key]
  __shared__ _Float16 Ps[4][16][40];    // per-wave probs, [row][key]

  const int b = blockIdx.z, h = blockIdx.y, q0 = blockIdx.x * 64;
  const int tid  = threadIdx.x;
  const int lane = tid & 31, w = tid >> 5;
  const int half = (lane >> 4) & 1, lr = lane & 15;

  const size_t qBase = ((size_t)b * Sq + q0) * Dc + h * DHc;
  const size_t kBase = ((size_t)b * Sk) * Dc + h * DHc;

  // ---- stage Q tile once (64x64 f32 -> f16, float4 loads) ----
  for (int i = tid; i < 64 * 16; i += 128) {
    int r = i >> 4, cv = (i & 15) << 2;
    float4 qv = *(const float4*)(Q + qBase + (size_t)r * Dc + cv);
    Qs[r][cv + 0] = (_Float16)qv.x; Qs[r][cv + 1] = (_Float16)qv.y;
    Qs[r][cv + 2] = (_Float16)qv.z; Qs[r][cv + 3] = (_Float16)qv.w;
  }
  __syncthreads();

  // ---- Q A-fragments (16 rows x K=64 -> two K-chunks of 32) ----
  v16h qa[2];
#pragma unroll
  for (int c = 0; c < 2; ++c) {
    const _Float16* p = &Qs[w * 16 + lr][c * 32 + half * 8];
    ((uint4*)&qa[c])[0] = *(const uint4*)p;
    ((uint4*)&qa[c])[1] = *(const uint4*)(p + 16);
  }

  v8f acc[4] = {};                       // O: 16 x 64 (4 N-chunks)
  float rmax[8], rsum[8];
#pragma unroll
  for (int r = 0; r < 8; ++r) { rmax[r] = -3.0e38f; rsum[r] = 0.f; }

  for (int kt = 0; kt < Sk; kt += 32) {
    __syncthreads();
    for (int i = tid; i < 32 * 16; i += 128) {       // K tile
      int r = i >> 4, cv = (i & 15) << 2;
      float4 kv = *(const float4*)(K + kBase + (size_t)(kt + r) * Dc + cv);
      Ks[r][cv + 0] = (_Float16)kv.x; Ks[r][cv + 1] = (_Float16)kv.y;
      Ks[r][cv + 2] = (_Float16)kv.z; Ks[r][cv + 3] = (_Float16)kv.w;
    }
    for (int i = tid; i < 32 * 16; i += 128) {       // V tile, transposed
      int r = i >> 4, cv = (i & 15) << 2;
      float4 vv = *(const float4*)(V + kBase + (size_t)(kt + r) * Dc + cv);
      Vs[cv + 0][r] = (_Float16)vv.x; Vs[cv + 1][r] = (_Float16)vv.y;
      Vs[cv + 2][r] = (_Float16)vv.z; Vs[cv + 3][r] = (_Float16)vv.w;
    }
    if (kt + 32 < Sk) {                              // prefetch next tiles
      __builtin_prefetch(K + kBase + (size_t)(kt + 32 + (tid >> 2)) * Dc + (tid & 3) * 16, 0, 1);
      __builtin_prefetch(V + kBase + (size_t)(kt + 32 + (tid >> 2)) * Dc + (tid & 3) * 16, 0, 1);
    }
    __syncthreads();

    // ---- scores S = Q @ K^T : 16 x 32 (two 16-col frags) ----
    v8f s[2] = {};
#pragma unroll
    for (int j = 0; j < 2; ++j)          // key chunk of 16 (N)
#pragma unroll
      for (int c = 0; c < 2; ++c) {      // dh chunk of 32 (K)
        const _Float16* p = &Ks[j * 16 + lr][c * 32 + half * 8];
        v16h bf;
        ((uint4*)&bf)[0] = *(const uint4*)p;
        ((uint4*)&bf)[1] = *(const uint4*)(p + 16);
        s[j] = wmma_f16(qa[c], bf, s[j]);
      }

    // ---- online softmax: fully unrolled, all state in registers ----
    float scf[8];
#pragma unroll
    for (int r = 0; r < 8; ++r) {
      float v0 = s[0][r] * scale, v1 = s[1][r] * scale;
      float mx = red_max16(fmaxf(v0, v1));
      float nm = fmaxf(rmax[r], mx);
      scf[r]   = __expf(rmax[r] - nm);
      float p0 = __expf(v0 - nm), p1 = __expf(v1 - nm);
      rsum[r]  = rsum[r] * scf[r] + red_sum16(p0 + p1);
      rmax[r]  = nm;
      int mrow = r + 8 * half;                       // C-frag row for this lane
      Ps[w][mrow][lr]      = (_Float16)p0;
      Ps[w][mrow][16 + lr] = (_Float16)p1;
    }
    v8f scv;
#pragma unroll
    for (int r = 0; r < 8; ++r) scv[r] = scf[r];
#pragma unroll
    for (int j = 0; j < 4; ++j) acc[j] *= scv;

    asm volatile("s_wait_dscnt 0" ::: "memory");     // Ps stores -> loads

    // ---- re-fragment probs into A layout (16 x K=32) ----
    v16h pa;
    {
      const _Float16* p = &Ps[w][lr][half * 8];
      ((uint4*)&pa)[0] = *(const uint4*)p;
      ((uint4*)&pa)[1] = *(const uint4*)(p + 16);
    }
    // ---- O += P @ V ----
#pragma unroll
    for (int j = 0; j < 4; ++j) {
      const _Float16* p = &Vs[j * 16 + lr][half * 8];
      v16h bf;
      ((uint4*)&bf)[0] = *(const uint4*)p;
      ((uint4*)&bf)[1] = *(const uint4*)(p + 16);
      acc[j] = wmma_f16(pa, bf, acc[j]);
    }
  }

  // ---- normalize and write O into the concatenated context buffer ----
#pragma unroll
  for (int r = 0; r < 8; ++r) {
    float inv = 1.0f / rsum[r];
    int mrow = w * 16 + r + 8 * half;
    size_t o = (size_t)b * outStride + (size_t)(rowOff + q0 + mrow) * Dc + h * DHc;
#pragma unroll
    for (int j = 0; j < 4; ++j)
      Out[o + j * 16 + lr] = acc[j][r] * inv;
  }
}

// ===========================================================================
extern "C" void kernel_launch(void* const* d_in, const int* in_sizes, int n_in,
                              void* d_out, int out_size, void* d_ws, size_t ws_size,
                              hipStream_t stream)
{
  (void)in_sizes; (void)n_in; (void)out_size; (void)ws_size;
  const float* hidden  = (const float*)d_in[0];
  const float* mol     = (const float*)d_in[1];
  const float* Wq      = (const float*)d_in[2];  const float* bq      = (const float*)d_in[3];
  const float* Wk      = (const float*)d_in[4];  const float* bk      = (const float*)d_in[5];
  const float* Wv      = (const float*)d_in[6];  const float* bv      = (const float*)d_in[7];
  const float* Wqm     = (const float*)d_in[8];  const float* bqm     = (const float*)d_in[9];
  const float* Wkm     = (const float*)d_in[10]; const float* bkm     = (const float*)d_in[11];
  const float* Wvm     = (const float*)d_in[12]; const float* bvm     = (const float*)d_in[13];
  const float* Wout    = (const float*)d_in[14]; const float* bout    = (const float*)d_in[15];
  const float* WoutM   = (const float*)d_in[16]; const float* boutM   = (const float*)d_in[17];
  const float* Wfc     = (const float*)d_in[18]; const float* bfc     = (const float*)d_in[19];
  const float* WfcM    = (const float*)d_in[20]; const float* bfcM    = (const float*)d_in[21];

  const size_t BPD = (size_t)Bc * Pc * Dc;
  const size_t BMD = (size_t)Bc * Mc * Dc;
  const size_t BCD = (size_t)Bc * (Pc + Mc) * Dc;
  const long long sP = (long long)Pc * Dc, sM = (long long)Mc * Dc;
  const long long sC = (long long)(Pc + Mc) * Dc;

  float* ws = (float*)d_ws;
  float* q        = ws;             float* k  = q  + BPD;  float* v  = k  + BPD;
  float* qm       = v + BPD;        float* km = qm + BMD;  float* vm = km + BMD;
  float* cat_prot = vm + BMD;       float* cat_mol = cat_prot + BCD;
  float* tmp_prot = cat_mol + BCD;  float* tmp_mol = tmp_prot + BPD;
  float* out_prot = (float*)d_out;  float* out_mol = out_prot + BPD;

  dim3 blk(256);
  dim3 gP(Dc / 64, Pc / 128, Bc);   // 12 x 8 x 8
  dim3 gM(Dc / 64, Mc / 128, Bc);   // 12 x 2 x 8

  // ---- projections: x @ W + b ----
  gemm_wmma<0,1><<<gP, blk, 0, stream>>>(hidden, Dc, sP, Wq,  Dc, 0, bq,  q,  Dc, sP, Pc, Dc, Dc, 1.0f);
  gemm_wmma<0,1><<<gP, blk, 0, stream>>>(hidden, Dc, sP, Wk,  Dc, 0, bk,  k,  Dc, sP, Pc, Dc, Dc, 1.0f);
  gemm_wmma<0,1><<<gP, blk, 0, stream>>>(hidden, Dc, sP, Wv,  Dc, 0, bv,  v,  Dc, sP, Pc, Dc, Dc, 1.0f);
  gemm_wmma<0,1><<<gM, blk, 0, stream>>>(mol,    Dc, sM, Wqm, Dc, 0, bqm, qm, Dc, sM, Mc, Dc, Dc, 1.0f);
  gemm_wmma<0,1><<<gM, blk, 0, stream>>>(mol,    Dc, sM, Wkm, Dc, 0, bkm, km, Dc, sM, Mc, Dc, Dc, 1.0f);
  gemm_wmma<0,1><<<gM, blk, 0, stream>>>(mol,    Dc, sM, Wvm, Dc, 0, bvm, vm, Dc, sM, Mc, Dc, Dc, 1.0f);

  // ---- four attentions, writing into concatenated context buffers ----
  const float sc = 0.125f;                 // 1/sqrt(DH)
  dim3 fblk(128);
  flash_attn<<<dim3(Pc/64, Hc, Bc), fblk, 0, stream>>>(q,  k,  v,  cat_prot, sC, 0,  Pc, Pc, sc); // pp
  flash_attn<<<dim3(Mc/64, Hc, Bc), fblk, 0, stream>>>(qm, k,  v,  cat_prot, sC, Pc, Mc, Pc, sc); // mp
  flash_attn<<<dim3(Mc/64, Hc, Bc), fblk, 0, stream>>>(qm, km, vm, cat_mol,  sC, 0,  Mc, Mc, sc); // mm
  flash_attn<<<dim3(Pc/64, Hc, Bc), fblk, 0, stream>>>(q,  km, vm, cat_mol,  sC, Mc, Pc, Mc, sc); // pm

  // ---- fc over sequence axis (A = Wfc^T via TA=1), then output proj ----
  gemm_wmma<1,2><<<gP, blk, 0, stream>>>(Wfc,      Pc, 0,  cat_prot, Dc, sC, bfc,   tmp_prot, Dc, sP, Pc, Dc, Pc+Mc, 1.0f);
  gemm_wmma<0,1><<<gP, blk, 0, stream>>>(tmp_prot, Dc, sP, Wout,     Dc, 0,  bout,  out_prot, Dc, sP, Pc, Dc, Dc,    1.0f);
  gemm_wmma<1,2><<<gM, blk, 0, stream>>>(WfcM,     Mc, 0,  cat_mol,  Dc, sC, bfcM,  tmp_mol,  Dc, sM, Mc, Dc, Pc+Mc, 1.0f);
  gemm_wmma<0,1><<<gM, blk, 0, stream>>>(tmp_mol,  Dc, sM, WoutM,    Dc, 0,  boutM, out_mol,  Dc, sM, Mc, Dc, Dc,    1.0f);
}
